// RWKV_2654289789468
// MI455X (gfx1250) — matrix operand order repaired and checked by reading, not compiled, not run on hardware
//
#include <hip/hip_runtime.h>
#include <hip/hip_bf16.h>

#define Bc 8
#define Tc 2048
#define Ec 1024
#define Hc 1024
#define Fc 4096
#define Mc (Bc * Tc)   // 16384 tokens

typedef __attribute__((ext_vector_type(16))) __bf16 v16bf;
typedef __attribute__((ext_vector_type(8)))  __bf16 v8bf;
typedef __attribute__((ext_vector_type(8)))  float  v8f;
typedef unsigned int v4u __attribute__((ext_vector_type(4)));
typedef int          v8i __attribute__((ext_vector_type(8)));
typedef int          v4i __attribute__((ext_vector_type(4)));

__device__ __forceinline__ float sigmoidf_(float x) {
    return 1.0f / (1.0f + __expf(-x));
}

// ---------------- Tensor Data Mover: 2D tile (global -> LDS) ----------------
// Builds D# group0 (128b) + group1 (256b) per CDNA5 ISA 8.3/8.4 and issues
// tensor_load_to_lds (TENSORcnt-tracked).  2-D tile: tile_dim0 x tile_dim1
// elements of 2 bytes, line stride = stride0 elements.
// Toolchain is the 6-arg (clang-23 / therock) form:
//   (uint32x4 g0, int32x8 g1, int32x4 g2, int32x4 g3, int32x8 g4, i32 cpol)
__device__ __forceinline__ void tdm_load_2d(unsigned long long gaddr,
                                            unsigned lds_addr,
                                            unsigned tensor_d0, unsigned tensor_d1,
                                            unsigned tile_w, unsigned tile_h,
                                            unsigned long long stride0) {
    v4u g0;
    g0[0] = 1u;                                   // count=1 (valid), no gather
    g0[1] = lds_addr;                             // LDS byte address
    g0[2] = (unsigned)(gaddr & 0xFFFFFFFFull);    // global_addr[31:0]
    g0[3] = (unsigned)((gaddr >> 32) & 0x01FFFFFFull) | (2u << 30);  // [56:32] | type=2
    v8i g1;
    g1[0] = (int)(1u << 16);                      // wg_mask=0, data_size=1 (2 bytes)
    g1[1] = (int)((tensor_d0 & 0xFFFFu) << 16);   // tensor_dim0[15:0] (bits 63:48)
    g1[2] = (int)((tensor_d0 >> 16) | ((tensor_d1 & 0xFFFFu) << 16));
    g1[3] = (int)((tensor_d1 >> 16) | (tile_w << 16));   // tile_dim0 (bits 127:112)
    g1[4] = (int)(tile_h & 0xFFFFu);              // tile_dim1; tile_dim2=0 (2D)
    g1[5] = (int)(stride0 & 0xFFFFFFFFull);       // tensor_dim0_stride[31:0]
    g1[6] = (int)((stride0 >> 32) & 0xFFFFull);   // tensor_dim0_stride[47:32]
    g1[7] = 0;
    v4i z4 = {0, 0, 0, 0};
    v8i z8 = {0, 0, 0, 0, 0, 0, 0, 0};
    __builtin_amdgcn_tensor_load_to_lds(g0, g1, z4, z4, z8, 0);
}

// ---------------- weight transpose + f32 -> bf16 convert ----------------
// src is (K, N) row-major f32; dst is (N, K) row-major bf16: dst[n*K+k] = src[k*N+n]
__global__ __launch_bounds__(256) void convtr_kernel(const float* __restrict__ src,
                                                     __bf16* __restrict__ dst,
                                                     int K, int N) {
    int idx = blockIdx.x * 256 + threadIdx.x;
    if (idx >= N * K) return;
    int n = idx / K;
    int k = idx - n * K;
    dst[idx] = (__bf16)src[(size_t)k * N + n];
}

// ---------------- LayerNorm over E=1024, one block per token row ----------------
__global__ __launch_bounds__(256) void ln_kernel(const float* __restrict__ x,
                                                 const float* __restrict__ g,
                                                 const float* __restrict__ b,
                                                 __bf16* __restrict__ out) {
    __shared__ float sa[256], sb[256];
    const int tid = threadIdx.x;
    const size_t base = (size_t)blockIdx.x * Ec;
    float s = 0.f, s2 = 0.f;
    #pragma unroll
    for (int i = tid; i < Ec; i += 256) {
        float v = x[base + i];
        s += v; s2 += v * v;
    }
    sa[tid] = s; sb[tid] = s2;
    __syncthreads();
    for (int st = 128; st > 0; st >>= 1) {
        if (tid < st) { sa[tid] += sa[tid + st]; sb[tid] += sb[tid + st]; }
        __syncthreads();
    }
    float mean = sa[0] * (1.0f / Ec);
    float var  = sb[0] * (1.0f / Ec) - mean * mean;
    float rstd = rsqrtf(var + 1e-3f);
    #pragma unroll
    for (int i = tid; i < Ec; i += 256) {
        out[base + i] = (__bf16)((x[base + i] - mean) * rstd * g[i] + b[i]);
    }
}

// ---------------- time-shift mix (3 outputs for time-mixing) ----------------
__global__ __launch_bounds__(256) void mix3_kernel(const __bf16* __restrict__ h,
                                                   const float* __restrict__ mk,
                                                   const float* __restrict__ mv,
                                                   const float* __restrict__ mr,
                                                   __bf16* __restrict__ xk,
                                                   __bf16* __restrict__ xv,
                                                   __bf16* __restrict__ xr) {
    size_t idx = (size_t)blockIdx.x * 256 + threadIdx.x;
    if (idx >= (size_t)Mc * Ec) return;
    int e  = (int)(idx & (Ec - 1));
    int bt = (int)(idx >> 10);
    int t  = bt & (Tc - 1);
    float hv = (float)h[idx];
    float h1 = (t > 0) ? (float)h[idx - Ec] : 0.0f;
    float a;
    a = mk[e]; xk[idx] = (__bf16)(hv * a + h1 * (1.0f - a));
    a = mv[e]; xv[idx] = (__bf16)(hv * a + h1 * (1.0f - a));
    a = mr[e]; xr[idx] = (__bf16)(hv * a + h1 * (1.0f - a));
}

// ---------------- time-shift mix (2 outputs for channel-mixing) ----------------
__global__ __launch_bounds__(256) void mix2_kernel(const __bf16* __restrict__ h,
                                                   const float* __restrict__ mk,
                                                   const float* __restrict__ mr,
                                                   __bf16* __restrict__ xk,
                                                   __bf16* __restrict__ xr) {
    size_t idx = (size_t)blockIdx.x * 256 + threadIdx.x;
    if (idx >= (size_t)Mc * Ec) return;
    int e  = (int)(idx & (Ec - 1));
    int bt = (int)(idx >> 10);
    int t  = bt & (Tc - 1);
    float hv = (float)h[idx];
    float h1 = (t > 0) ? (float)h[idx - Ec] : 0.0f;
    float a;
    a = mk[e]; xk[idx] = (__bf16)(hv * a + h1 * (1.0f - a));
    a = mr[e]; xr[idx] = (__bf16)(hv * a + h1 * (1.0f - a));
}

// ---------------- WKV recurrence: one thread per (b, h) channel ----------------
__global__ __launch_bounds__(256) void wkv_kernel(const float* __restrict__ kf,
                                                  const float* __restrict__ vf,
                                                  const float* __restrict__ td,
                                                  const float* __restrict__ tf,
                                                  float* __restrict__ wkv) {
    int c = blockIdx.x * 256 + threadIdx.x;
    if (c >= Bc * Hc) return;
    int b  = c / Hc;
    int hh = c - b * Hc;
    float w = -__expf(td[hh]);
    float u = tf[hh];
    float aa = 0.f, bb = 0.f, pp = -1e38f;
    size_t base = (size_t)b * Tc * Hc + hh;
    for (int t = 0; t < Tc; ++t) {
        size_t i = base + (size_t)t * Hc;
        float kt = kf[i];
        float vt = vf[i];
        float ww = u + kt;
        float q  = fmaxf(pp, ww);
        float e1 = __expf(pp - q);
        float e2 = __expf(ww - q);
        wkv[i] = (e1 * aa + e2 * vt) / (e1 * bb + e2);
        float ww2 = pp + w;
        float q2  = fmaxf(ww2, kt);
        float e1b = __expf(ww2 - q2);
        float e2b = __expf(kt - q2);
        aa = e1b * aa + e2b * vt;
        bb = e1b * bb + e2b;
        pp = q2;
    }
}

// ---------------- sigmoid(r) * wkv -> bf16 ----------------
__global__ __launch_bounds__(256) void sigmul_kernel(const float* __restrict__ r,
                                                     const float* __restrict__ wkv,
                                                     __bf16* __restrict__ out) {
    size_t idx = (size_t)blockIdx.x * 256 + threadIdx.x;
    if (idx >= (size_t)Mc * Hc) return;
    out[idx] = (__bf16)(sigmoidf_(r[idx]) * wkv[idx]);
}

// ---------------- TDM-staged bf16 WMMA GEMM: C(MxN) = A(MxK) * Bt(NxK)^T ----
// Block tile 128x64, 8 waves.  Per K-step (BK=32): wave 0 DMAs the A tile
// (128x32) and B tile (64x32) into LDS via tensor_load_to_lds, waits on
// TENSORcnt, barrier, then each wave computes a 16x64 strip (4 WMMAs) from
// LDS (ds_load_b128 fragment reads).
// EP 0: Cf = acc                         (raw f32)
// EP 1: Cb = bf16(relu(acc)^2)           (channel-mix squared relu)
// EP 2: Cf = addv + acc                  (residual add)
// EP 3: Cf = addv + sigmoid(gate) * acc  (final gated residual)
template <int EP>
__global__ __launch_bounds__(256) void gemm_wmma_tdm(const __bf16* __restrict__ A,
                                                     const __bf16* __restrict__ Bt,
                                                     float* __restrict__ Cf,
                                                     __bf16* __restrict__ Cb,
                                                     const float* __restrict__ addv,
                                                     const float* __restrict__ gate,
                                                     int M, int N, int K) {
    __shared__ __bf16 As[128 * 32];   // 8 KB
    __shared__ __bf16 Bs[64 * 32];    // 4 KB
    const int lane = threadIdx.x & 31;
    const int wave = threadIdx.x >> 5;
    const int half = lane >> 4;
    const int l16  = lane & 15;
    const int m0 = blockIdx.y * 128;
    const int n0 = blockIdx.x * 64;

    const unsigned ldsA = (unsigned)(size_t)(void*)As;
    const unsigned ldsB = (unsigned)(size_t)(void*)Bs;
    const unsigned long long baseA = (unsigned long long)(size_t)A + ((size_t)m0 * K) * 2ull;
    const unsigned long long baseB = (unsigned long long)(size_t)Bt + ((size_t)n0 * K) * 2ull;

    // LDS fragment pointers (bf16 ISA layouts => contiguous 16B chunks)
    const __bf16* pa = As + (wave * 16 + l16) * 32;
    v8f acc0 = {}, acc1 = {}, acc2 = {}, acc3 = {};

    for (int kk = 0; kk < K; kk += 32) {
        if (threadIdx.x < 32) {   // wave 0 issues the DMAs and waits on them
            tdm_load_2d(baseA + (size_t)kk * 2ull, ldsA,
                        (unsigned)K, (unsigned)M, 32u, 128u, (unsigned long long)K);
            tdm_load_2d(baseB + (size_t)kk * 2ull, ldsB,
                        (unsigned)K, (unsigned)N, 32u, 64u, (unsigned long long)K);
            __builtin_amdgcn_s_wait_tensorcnt(0);
        }
        __syncthreads();
        // A fragment: elements 0-7 -> K = half*8 + 0..7 ; 8-15 -> K = 16 + half*8 + 0..7
        v8bf a0 = *(const v8bf*)(pa + half * 8);
        v8bf a1 = *(const v8bf*)(pa + 16 + half * 8);
        v16bf av = __builtin_shufflevector(a0, a1, 0,1,2,3,4,5,6,7,8,9,10,11,12,13,14,15);
        #pragma unroll
        for (int ns = 0; ns < 4; ++ns) {
            const __bf16* pb = Bs + (ns * 16 + l16) * 32;
            v8bf b0 = *(const v8bf*)(pb + half * 8);
            v8bf b1 = *(const v8bf*)(pb + 16 + half * 8);
            v16bf bv = __builtin_shufflevector(b0, b1, 0,1,2,3,4,5,6,7,8,9,10,11,12,13,14,15);
            if (ns == 0) acc0 = __builtin_amdgcn_wmma_f32_16x16x32_bf16(false, av, false, bv, (short)0, acc0, false, false);
            if (ns == 1) acc1 = __builtin_amdgcn_wmma_f32_16x16x32_bf16(false, av, false, bv, (short)0, acc1, false, false);
            if (ns == 2) acc2 = __builtin_amdgcn_wmma_f32_16x16x32_bf16(false, av, false, bv, (short)0, acc2, false, false);
            if (ns == 3) acc3 = __builtin_amdgcn_wmma_f32_16x16x32_bf16(false, av, false, bv, (short)0, acc3, false, false);
        }
        __syncthreads();
    }

    #pragma unroll
    for (int ns = 0; ns < 4; ++ns) {
        v8f acc = (ns == 0) ? acc0 : (ns == 1) ? acc1 : (ns == 2) ? acc2 : acc3;
        int col = n0 + ns * 16 + l16;
        #pragma unroll
        for (int r = 0; r < 8; ++r) {
            int row = m0 + wave * 16 + half * 8 + r;   // C layout: lanes16-31 = M+8
            size_t ci = (size_t)row * N + col;
            float v = acc[r];
            if (EP == 0) {
                Cf[ci] = v;
            } else if (EP == 1) {
                float z = v > 0.f ? v : 0.f;
                Cb[ci] = (__bf16)(z * z);
            } else if (EP == 2) {
                Cf[ci] = addv[ci] + v;
            } else {
                Cf[ci] = addv[ci] + sigmoidf_(gate[ci]) * v;
            }
        }
    }
}

// =======================================================================
extern "C" void kernel_launch(void* const* d_in, const int* in_sizes, int n_in,
                              void* d_out, int out_size, void* d_ws, size_t ws_size,
                              hipStream_t stream) {
    const float* inputs   = (const float*)d_in[0];
    const float* ln1_g    = (const float*)d_in[1];
    const float* ln1_b    = (const float*)d_in[2];
    const float* ln2_g    = (const float*)d_in[3];
    const float* ln2_b    = (const float*)d_in[4];
    const float* tm_mix_k = (const float*)d_in[5];
    const float* tm_mix_v = (const float*)d_in[6];
    const float* tm_mix_r = (const float*)d_in[7];
    const float* tm_Wk    = (const float*)d_in[8];
    const float* tm_Wv    = (const float*)d_in[9];
    const float* tm_Wr    = (const float*)d_in[10];
    const float* tdecay   = (const float*)d_in[11];
    const float* tfirst   = (const float*)d_in[12];
    const float* Wo       = (const float*)d_in[13];
    const float* cm_mix_k = (const float*)d_in[14];
    const float* cm_mix_r = (const float*)d_in[15];
    const float* cm_Wk    = (const float*)d_in[16];
    const float* cm_Wv    = (const float*)d_in[17];
    const float* cm_Wr    = (const float*)d_in[18];
    float* out = (float*)d_out;

    // ---- workspace layout ----
    char* p = (char*)d_ws;
    auto take = [&](size_t bytes) { char* r = p; p += (bytes + 255) & ~(size_t)255; return r; };
    __bf16* WkT   = (__bf16*)take(sizeof(__bf16) * (size_t)Hc * Ec);
    __bf16* WvT   = (__bf16*)take(sizeof(__bf16) * (size_t)Hc * Ec);
    __bf16* WrT   = (__bf16*)take(sizeof(__bf16) * (size_t)Hc * Ec);
    __bf16* WoT   = (__bf16*)take(sizeof(__bf16) * (size_t)Ec * Hc);
    __bf16* cmWkT = (__bf16*)take(sizeof(__bf16) * (size_t)Fc * Ec);
    __bf16* cmWvT = (__bf16*)take(sizeof(__bf16) * (size_t)Ec * Fc);
    __bf16* cmWrT = (__bf16*)take(sizeof(__bf16) * (size_t)Ec * Ec);
    __bf16* hb    = (__bf16*)take(sizeof(__bf16) * (size_t)Mc * Ec);
    __bf16* xk    = (__bf16*)take(sizeof(__bf16) * (size_t)Mc * Ec);
    __bf16* xr    = (__bf16*)take(sizeof(__bf16) * (size_t)Mc * Ec);
    __bf16* xv    = (__bf16*)take(sizeof(__bf16) * (size_t)Mc * Ec);
    float*  kf    = (float*) take(sizeof(float)  * (size_t)Mc * Hc);
    float*  vf    = (float*) take(sizeof(float)  * (size_t)Mc * Hc);
    float*  rf    = (float*) take(sizeof(float)  * (size_t)Mc * Hc);
    float*  wkvf  = (float*) take(sizeof(float)  * (size_t)Mc * Hc);
    float*  x1    = (float*) take(sizeof(float)  * (size_t)Mc * Ec);
    // Aliases (lifetimes do not overlap):
    __bf16* rwkvb = xv;                 // sigmoid(r)*wkv, bf16      (xv dead after v GEMM)
    __bf16* kcb   = (__bf16*)kf;        // relu(xk2@cmWk)^2, 134 MB  (k/v/r dead after wkv+rwkv)
    float*  rcf   = wkvf;               // xr2@cmWr raw              (wkv dead after rwkvb)

    const int ELT = 256;
    // ---- 1. convert + transpose all weights to bf16 ----
    convtr_kernel<<<(Ec * Hc) / ELT, ELT, 0, stream>>>(tm_Wk, WkT,   Ec, Hc);
    convtr_kernel<<<(Ec * Hc) / ELT, ELT, 0, stream>>>(tm_Wv, WvT,   Ec, Hc);
    convtr_kernel<<<(Ec * Hc) / ELT, ELT, 0, stream>>>(tm_Wr, WrT,   Ec, Hc);
    convtr_kernel<<<(Hc * Ec) / ELT, ELT, 0, stream>>>(Wo,    WoT,   Hc, Ec);
    convtr_kernel<<<(Ec * Fc) / ELT, ELT, 0, stream>>>(cm_Wk, cmWkT, Ec, Fc);
    convtr_kernel<<<(Fc * Ec) / ELT, ELT, 0, stream>>>(cm_Wv, cmWvT, Fc, Ec);
    convtr_kernel<<<(Ec * Ec) / ELT, ELT, 0, stream>>>(cm_Wr, cmWrT, Ec, Ec);

    // ---- 2. time-mixing front end ----
    ln_kernel<<<Mc, 256, 0, stream>>>(inputs, ln1_g, ln1_b, hb);
    mix3_kernel<<<((size_t)Mc * Ec) / ELT, ELT, 0, stream>>>(hb, tm_mix_k, tm_mix_v, tm_mix_r,
                                                             xk, xv, xr);
    // ---- 3. k, v, r GEMMs (M=16384, N=1024, K=1024) ----
    {
        dim3 g(Hc / 64, Mc / 128);
        gemm_wmma_tdm<0><<<g, 256, 0, stream>>>(xk, WkT, kf, nullptr, nullptr, nullptr, Mc, Hc, Ec);
        gemm_wmma_tdm<0><<<g, 256, 0, stream>>>(xv, WvT, vf, nullptr, nullptr, nullptr, Mc, Hc, Ec);
        gemm_wmma_tdm<0><<<g, 256, 0, stream>>>(xr, WrT, rf, nullptr, nullptr, nullptr, Mc, Hc, Ec);
    }
    // ---- 4. WKV scan: 8192 channels ----
    wkv_kernel<<<(Bc * Hc) / ELT, ELT, 0, stream>>>(kf, vf, tdecay, tfirst, wkvf);
    // ---- 5. sigmoid(r) * wkv -> bf16, then attn out + residual into x1 ----
    sigmul_kernel<<<((size_t)Mc * Hc) / ELT, ELT, 0, stream>>>(rf, wkvf, rwkvb);
    gemm_wmma_tdm<2><<<dim3(Ec / 64, Mc / 128), 256, 0, stream>>>(
        rwkvb, WoT, x1, nullptr, inputs, nullptr, Mc, Ec, Hc);

    // ---- 6. channel-mixing ----
    ln_kernel<<<Mc, 256, 0, stream>>>(x1, ln2_g, ln2_b, hb);
    mix2_kernel<<<((size_t)Mc * Ec) / ELT, ELT, 0, stream>>>(hb, cm_mix_k, cm_mix_r, xk, xr);
    // kc = relu(xk @ cmWk)^2 stored bf16 (N=4096)
    gemm_wmma_tdm<1><<<dim3(Fc / 64, Mc / 128), 256, 0, stream>>>(
        xk, cmWkT, nullptr, kcb, nullptr, nullptr, Mc, Fc, Ec);
    // rc_raw = xr @ cmWr (sigmoid applied in the final epilogue)
    gemm_wmma_tdm<0><<<dim3(Ec / 64, Mc / 128), 256, 0, stream>>>(
        xr, cmWrT, rcf, nullptr, nullptr, nullptr, Mc, Ec, Ec);
    // out = x1 + sigmoid(rc) * (kc @ cmWv)   (K=4096)
    gemm_wmma_tdm<3><<<dim3(Ec / 64, Mc / 128), 256, 0, stream>>>(
        kcb, cmWvT, out, nullptr, x1, rcf, Mc, Ec, Fc);
    (void)in_sizes; (void)n_in; (void)out_size; (void)ws_size;
}
